// Double_Decoder_MAE_All_attribute_30193620091451
// MI455X (gfx1250) — compile-verified, matrix operand order checked by reading, not run on hardware
//
#include <hip/hip_runtime.h>

#define B_ 8
#define N_ 8192
#define A_ 10
#define G_ 512
#define M_ 32
#define R_ (B_*G_*M_)   // 131072 rows through the MLPs
#define EPSF 1e-5f

typedef __attribute__((ext_vector_type(16))) _Float16 v16h;
typedef __attribute__((ext_vector_type(8)))  _Float16 v8h;
typedef __attribute__((ext_vector_type(8)))  float    v8f;

static __device__ __forceinline__ void put8(v16h& f, int e0, float4 u, float4 v)
{
    f[e0 + 0] = (_Float16)u.x; f[e0 + 1] = (_Float16)u.y;
    f[e0 + 2] = (_Float16)u.z; f[e0 + 3] = (_Float16)u.w;
    f[e0 + 4] = (_Float16)v.x; f[e0 + 5] = (_Float16)v.y;
    f[e0 + 6] = (_Float16)v.z; f[e0 + 7] = (_Float16)v.w;
}

static __device__ __forceinline__ float4 bnrelu4(float4 x, float4 s, float4 t)
{
    float4 r;
    r.x = fmaxf(x.x * s.x + t.x, 0.f);
    r.y = fmaxf(x.y * s.y + t.y, 0.f);
    r.z = fmaxf(x.z * s.z + t.z, 0.f);
    r.w = fmaxf(x.w * s.w + t.w, 0.f);
    return r;
}

// ---------------------------------------------------------------------------
// GEMM:  Y[r,c] = act(X[r,:]) . Wh[c,:] + bias[c]
//   X: f32, row stride K (K mult of 32);  Wh: f16, pre-padded, row stride K
//   act = FUSE ? relu(x*in_scale[k]+in_shift[k]) : x      (fused BN+ReLU)
// Fused per-channel sum/sumsq of pre-BN outputs for the next BatchNorm.
// Wave tile = 16 rows x (16*COLT) cols: 1 A fragment feeds COLT WMMAs.
// Block = 8 row-waves = 128 rows x 16*COLT cols.  Grid = (C/(16*COLT), R/128).
// K, C compile-time -> every load/store offset folds to an immediate.
// ---------------------------------------------------------------------------
template<int K, int C, int COLT, bool FUSE>
__global__ __launch_bounds__(256) void k_gemm_v(
    const float* __restrict__ X, const _Float16* __restrict__ Wh,
    const float* __restrict__ bias, float* __restrict__ Y,
    const float* __restrict__ in_scale, const float* __restrict__ in_shift,
    float* __restrict__ o_sum, float* __restrict__ o_ssq)
{
    __shared__ float lsum[16 * COLT];
    __shared__ float lssq[16 * COLT];
    const int tid = threadIdx.x;
    for (int i = tid; i < 16 * COLT; i += 256) { lsum[i] = 0.f; lssq[i] = 0.f; }
    __syncthreads();

    const int wave = tid >> 5, lane = tid & 31;
    const int r0 = blockIdx.y * 128 + wave * 16;
    const int c0 = blockIdx.x * (16 * COLT);
    const int m  = r0 + (lane & 15);
    const int kb = (lane < 16) ? 0 : 8;   // per-lane-half K sub-base (ISA 7.12.2)

    const float*    xrow  = X  + (size_t)m * K + kb;
    const _Float16* wbase = Wh + (size_t)(c0 + (lane & 15)) * K + kb;

    v8f acc[COLT];
#pragma unroll
    for (int ct = 0; ct < COLT; ++ct) acc[ct] = (v8f){};

    for (int kk = 0; kk < K; kk += 32) {
        // A fragment: k = kb..kb+7 (elems 0-7), kb+16..kb+23 (elems 8-15)
        float4 xa = *(const float4*)(xrow + kk);
        float4 xb = *(const float4*)(xrow + kk + 4);
        float4 xc = *(const float4*)(xrow + kk + 16);
        float4 xd = *(const float4*)(xrow + kk + 20);
        if (FUSE) {
            const float* sp = in_scale + kk + kb;
            const float* tp = in_shift + kk + kb;
            xa = bnrelu4(xa, *(const float4*)(sp),      *(const float4*)(tp));
            xb = bnrelu4(xb, *(const float4*)(sp + 4),  *(const float4*)(tp + 4));
            xc = bnrelu4(xc, *(const float4*)(sp + 16), *(const float4*)(tp + 16));
            xd = bnrelu4(xd, *(const float4*)(sp + 20), *(const float4*)(tp + 20));
        }
        v16h a;
        put8(a, 0, xa, xb);
        put8(a, 8, xc, xd);
#pragma unroll
        for (int ct = 0; ct < COLT; ++ct) {
            const _Float16* wp = wbase + (size_t)ct * 16 * K + kk;
            const v8h lo = *(const v8h*)(wp);        // k = kb..kb+7   (16B)
            const v8h hi = *(const v8h*)(wp + 16);   // k = kb+16..+23 (16B)
            const v16h b = __builtin_shufflevector(lo, hi,
                0, 1, 2, 3, 4, 5, 6, 7, 8, 9, 10, 11, 12, 13, 14, 15);
            acc[ct] = __builtin_amdgcn_wmma_f32_16x16x32_f16(false, a, false, b,
                                                             (short)0, acc[ct],
                                                             false, false);
        }
    }

    // Epilogue: bias, store pre-BN Y, per-channel sum/sumsq for next BN.
    const int rb = r0 + ((lane < 16) ? 0 : 8);   // C/D: VGPR i -> row rb+i
#pragma unroll
    for (int ct = 0; ct < COLT; ++ct) {
        const int n = c0 + ct * 16 + (lane & 15);
        const float bs = bias[n];
        float* yb = Y + (size_t)rb * C + n;
        float s = 0.f, q = 0.f;
#pragma unroll
        for (int i = 0; i < 8; ++i) {
            const float y = acc[ct][i] + bs;
            yb[(size_t)i * C] = y;               // immediate offsets (C const)
            s += y; q += y * y;
        }
        atomicAdd(&lsum[ct * 16 + (lane & 15)], s);   // ds_add_f32
        atomicAdd(&lssq[ct * 16 + (lane & 15)], q);
    }
    __syncthreads();
    for (int i = tid; i < 16 * COLT; i += 256) {
        atomicAdd(&o_sum[c0 + i], lsum[i]);
        atomicAdd(&o_ssq[c0 + i], lssq[i]);
    }
}

// Convert weight (C,K) f32 -> (C,Kp) f16, zero-padded in K. One-time, tiny.
__global__ void k_cvtw(const float* __restrict__ w, _Float16* __restrict__ wh,
                       int C, int K, int Kp)
{
    const int i = blockIdx.x * blockDim.x + threadIdx.x;
    if (i >= C * Kp) return;
    const int c = i / Kp, k = i % Kp;
    wh[i] = (k < K) ? (_Float16)w[c * K + k] : (_Float16)0.f;
}

// ---------------------------------------------------------------------------
// Farthest point sampling: one block per batch; dist[] lives in LDS.
// ---------------------------------------------------------------------------
__global__ __launch_bounds__(256) void k_fps(const float* __restrict__ xyz,
                                             float* __restrict__ cen)
{
    __shared__ float dist[N_];
    __shared__ float rbest[256];
    __shared__ int   ridx[256];
    __shared__ float sc[3];
    __shared__ int   sfar;
    const int b = blockIdx.x, t = threadIdx.x;
    const float* xb = xyz + (size_t)b * N_ * A_;
    for (int nn = t; nn < N_; nn += 256) dist[nn] = 1e10f;
    if (t == 0) sfar = 0;
    __syncthreads();
    for (int g = 0; g < G_; ++g) {
        const int far = sfar;
        if (t < A_) cen[((size_t)b * G_ + g) * A_ + t] = xb[(size_t)far * A_ + t];
        if (t < 3)  sc[t] = xb[(size_t)far * A_ + t];
        __syncthreads();
        float bv = -1.f; int bi = N_;
        for (int j = 0; j < N_ / 256; ++j) {
            const int nn = t + j * 256;
            const float dx = xb[(size_t)nn * A_ + 0] - sc[0];
            const float dy = xb[(size_t)nn * A_ + 1] - sc[1];
            const float dz = xb[(size_t)nn * A_ + 2] - sc[2];
            const float d  = dx * dx + dy * dy + dz * dz;
            const float dn = fminf(dist[nn], d);
            dist[nn] = dn;
            if (dn > bv) { bv = dn; bi = nn; }
        }
        rbest[t] = bv; ridx[t] = bi;
        __syncthreads();
        for (int s = 128; s > 0; s >>= 1) {
            if (t < s) {
                const float v2 = rbest[t + s]; const int i2 = ridx[t + s];
                if (v2 > rbest[t] || (v2 == rbest[t] && i2 < ridx[t])) {
                    rbest[t] = v2; ridx[t] = i2;
                }
            }
            __syncthreads();
        }
        if (t == 0) sfar = ridx[0];
        __syncthreads();
    }
}

// ---------------------------------------------------------------------------
// kNN grouping: one block per (b,g); full 10-D distance, top-32 smallest via
// 32 LDS argmin passes; writes recentered pg rows padded to stride 32.
// ---------------------------------------------------------------------------
__global__ __launch_bounds__(256) void k_group(const float* __restrict__ xyz,
                                               const float* __restrict__ cen,
                                               float* __restrict__ pg)
{
    __shared__ float sd[N_];
    __shared__ float rbest[256];
    __shared__ int   ridx[256];
    __shared__ float sc[A_];
    __shared__ int   sel[M_];
    const int bg = blockIdx.x;
    const int b  = bg / G_;
    const int t  = threadIdx.x;
    const float* xb = xyz + (size_t)b * N_ * A_;
    if (t < A_) sc[t] = cen[(size_t)bg * A_ + t];
    __syncthreads();
    for (int nn = t; nn < N_; nn += 256) {
        float d = 0.f;
#pragma unroll
        for (int a = 0; a < A_; ++a) {
            const float df = xb[(size_t)nn * A_ + a] - sc[a];
            d += df * df;
        }
        sd[nn] = d;
    }
    __syncthreads();
    for (int m = 0; m < M_; ++m) {
        float bv = 3.4e38f; int bi = 0;
        for (int nn = t; nn < N_; nn += 256) {
            const float d = sd[nn];
            if (d < bv) { bv = d; bi = nn; }
        }
        rbest[t] = bv; ridx[t] = bi;
        __syncthreads();
        for (int s = 128; s > 0; s >>= 1) {
            if (t < s) {
                const float v2 = rbest[t + s]; const int i2 = ridx[t + s];
                if (v2 < rbest[t] || (v2 == rbest[t] && i2 < ridx[t])) {
                    rbest[t] = v2; ridx[t] = i2;
                }
            }
            __syncthreads();
        }
        if (t == 0) { sel[m] = ridx[0]; sd[ridx[0]] = 3.4e38f; }
        __syncthreads();
    }
    for (int i = t; i < M_ * 32; i += 256) {     // stride-32 padded rows
        const int m = i >> 5, a = i & 31;
        float v = 0.f;
        if (a < A_) {
            v = xb[(size_t)sel[m] * A_ + a];
            if (a < 3) v -= sc[a];
        }
        pg[(size_t)bg * M_ * 32 + i] = v;
    }
}

// BN finalize: scale = g*rsqrt(var+eps), shift = be - mean*scale
__global__ void k_bnfin(const float* __restrict__ sum, const float* __restrict__ ssq,
                        const float* __restrict__ g, const float* __restrict__ be,
                        float* __restrict__ scale, float* __restrict__ shift,
                        int C, float invR)
{
    const int c = blockIdx.x * blockDim.x + threadIdx.x;
    if (c >= C) return;
    const float m = sum[c] * invR;
    const float v = ssq[c] * invR - m * m;
    const float s = g[c] * rsqrtf(v + EPSF);
    scale[c] = s;
    shift[c] = be[c] - m * s;
}

__global__ void k_zero(float* __restrict__ p, int n)
{
    const int i = blockIdx.x * blockDim.x + threadIdx.x;
    if (i < n) p[i] = 0.f;
}

// T-Net maxpool over M with fused BN+ReLU
__global__ void k_tnet_maxpool(const float* __restrict__ Y, const float* __restrict__ scale,
                               const float* __restrict__ shift, float* __restrict__ hmax)
{
    const int i = blockIdx.x * blockDim.x + threadIdx.x;
    if (i >= 4096 * 512) return;
    const int g = i >> 9, c = i & 511;
    float mx = -3.4e38f;
    for (int m = 0; m < M_; ++m) {
        float y = Y[((size_t)g * M_ + m) * 512 + c] * scale[c] + shift[c];
        y = y > 0.f ? y : 0.f;
        mx = fmaxf(mx, y);
    }
    hmax[i] = mx;
}

// T-Net final 512->36 + bias + identity
__global__ void k_tmat(const float* __restrict__ hmax, const float* __restrict__ fw,
                       const float* __restrict__ fb, float* __restrict__ tmat)
{
    const int i = blockIdx.x * blockDim.x + threadIdx.x;
    if (i >= 4096 * 36) return;
    const int g = i / 36, j = i % 36;
    const float* h = hmax + (size_t)g * 512;
    const float* w = fw + (size_t)j * 512;
    float acc = 0.f;
    for (int k = 0; k < 512; ++k) acc += h[k] * w[k];
    acc += fb[j];
    if (j % 7 == 0) acc += 1.f;   // + eye(6).reshape(36)
    tmat[i] = acc;
}

// Apply 6x6 transform to first 6 channels of pg (stride-32 rows, in place)
__global__ void k_transform(float* __restrict__ pg, const float* __restrict__ tmat)
{
    const int i = blockIdx.x * blockDim.x + threadIdx.x;
    if (i >= R_) return;
    const int g = i / M_;
    float* row = pg + (size_t)i * 32;
    const float* t = tmat + (size_t)g * 36;
    float x[6];
#pragma unroll
    for (int c = 0; c < 6; ++c) x[c] = row[c];
#pragma unroll
    for (int d = 0; d < 6; ++d) {
        float a = 0.f;
#pragma unroll
        for (int c = 0; c < 6; ++c) a += x[c] * t[c * 6 + d];
        row[d] = a;
    }
}

// Final BN (no relu) + max over M -> d_out (4096 x 1024)
__global__ void k_final(const float* __restrict__ Y, const float* __restrict__ scale,
                        const float* __restrict__ shift, float* __restrict__ out)
{
    const int i = blockIdx.x * blockDim.x + threadIdx.x;
    if (i >= 4096 * 1024) return;
    const int g = i >> 10, c = i & 1023;
    float mx = -3.4e38f;
    for (int m = 0; m < M_; ++m) {
        const float y = Y[((size_t)g * M_ + m) * 1024 + c] * scale[c] + shift[c];
        mx = fmaxf(mx, y);
    }
    out[i] = mx;
}

// ---------------------------------------------------------------------------
extern "C" void kernel_launch(void* const* d_in, const int* in_sizes, int n_in,
                              void* d_out, int out_size, void* d_ws, size_t ws_size,
                              hipStream_t stream)
{
    const float* xyz    = (const float*)d_in[0];
    const float* tn_w1  = (const float*)d_in[1];
    const float* tn_b1  = (const float*)d_in[2];
    const float* tn_g1  = (const float*)d_in[3];
    const float* tn_be1 = (const float*)d_in[4];
    const float* tn_w2  = (const float*)d_in[5];
    const float* tn_b2  = (const float*)d_in[6];
    const float* tn_g2  = (const float*)d_in[7];
    const float* tn_be2 = (const float*)d_in[8];
    const float* tn_fw  = (const float*)d_in[9];
    const float* tn_fb  = (const float*)d_in[10];
    const float* w1 = (const float*)d_in[11]; const float* b1 = (const float*)d_in[12];
    const float* g1 = (const float*)d_in[13]; const float* be1= (const float*)d_in[14];
    const float* w2 = (const float*)d_in[15]; const float* b2 = (const float*)d_in[16];
    const float* g2 = (const float*)d_in[17]; const float* be2= (const float*)d_in[18];
    const float* w3 = (const float*)d_in[19]; const float* b3 = (const float*)d_in[20];
    const float* g3 = (const float*)d_in[21]; const float* be3= (const float*)d_in[22];
    const float* w4 = (const float*)d_in[23]; const float* b4 = (const float*)d_in[24];
    const float* g4 = (const float*)d_in[25]; const float* be4= (const float*)d_in[26];
    (void)in_sizes; (void)n_in; (void)out_size; (void)ws_size;

    float* ws = (float*)d_ws;
    size_t off = 0;
    auto carve = [&](size_t n) { float* p = ws + off; off += (n + 63) & ~(size_t)63; return p; };
    float* cen   = carve((size_t)B_ * G_ * A_);    // FPS centroids, all 10 attrs
    float* pg    = carve((size_t)R_ * 32);         // grouped points, rows padded to 32
    float* hmax  = carve((size_t)4096 * 512);      // T-Net pooled features
    float* tmat  = carve((size_t)4096 * 36);       // 6x6 transforms
    float* stats = carve(2 * 2496);                // sum+ssq, all 6 BN layers
    float* scsh  = carve(2 * 2496);                // scale+shift, all 6 BN layers
    // f16 weight copies (zero-padded K for the two small layers)
    _Float16* htw1 = (_Float16*)carve(64   * 32  / 2);
    _Float16* htw2 = (_Float16*)carve(512  * 64  / 2);
    _Float16* hw1  = (_Float16*)carve(128  * 32  / 2);
    _Float16* hw2  = (_Float16*)carve(256  * 128 / 2);
    _Float16* hw3  = (_Float16*)carve(512  * 256 / 2);
    _Float16* hw4  = (_Float16*)carve((size_t)1024 * 512 / 2);
    float* bufA  = carve((size_t)R_ * 1024);       // ping
    float* bufB  = carve((size_t)R_ * 1024);       // pong

    float* sum_tn1 = stats;          float* ssq_tn1 = sum_tn1 + 64;
    float* sum_tn2 = ssq_tn1 + 64;   float* ssq_tn2 = sum_tn2 + 512;
    float* sum_l1  = ssq_tn2 + 512;  float* ssq_l1  = sum_l1 + 128;
    float* sum_l2  = ssq_l1 + 128;   float* ssq_l2  = sum_l2 + 256;
    float* sum_l3  = ssq_l2 + 256;   float* ssq_l3  = sum_l3 + 512;
    float* sum_l4  = ssq_l3 + 512;   float* ssq_l4  = sum_l4 + 1024;
    float* sc_tn1 = scsh;            float* sh_tn1 = sc_tn1 + 64;
    float* sc_tn2 = sh_tn1 + 64;     float* sh_tn2 = sc_tn2 + 512;
    float* sc_l1  = sh_tn2 + 512;    float* sh_l1  = sc_l1 + 128;
    float* sc_l2  = sh_l1 + 128;     float* sh_l2  = sc_l2 + 256;
    float* sc_l3  = sh_l2 + 256;     float* sh_l3  = sc_l3 + 512;
    float* sc_l4  = sh_l3 + 512;     float* sh_l4  = sc_l4 + 1024;

    const float invR = 1.f / (float)R_;

    // 0) zero BN accumulators + one-time f16 weight conversion (zero-padded)
    k_zero<<<(2 * 2496 + 255) / 256, 256, 0, stream>>>(stats, 2 * 2496);
    k_cvtw<<<(64 * 32 + 255) / 256, 256, 0, stream>>>(tn_w1, htw1, 64, 6, 32);
    k_cvtw<<<(512 * 64 + 255) / 256, 256, 0, stream>>>(tn_w2, htw2, 512, 64, 64);
    k_cvtw<<<(128 * 32 + 255) / 256, 256, 0, stream>>>(w1, hw1, 128, 10, 32);
    k_cvtw<<<(256 * 128 + 255) / 256, 256, 0, stream>>>(w2, hw2, 256, 128, 128);
    k_cvtw<<<(512 * 256 + 255) / 256, 256, 0, stream>>>(w3, hw3, 512, 256, 256);
    k_cvtw<<<(1024 * 512 + 255) / 256, 256, 0, stream>>>(w4, hw4, 1024, 512, 512);

    // 1) FPS + grouping
    k_fps<<<B_, 256, 0, stream>>>(xyz, cen);
    k_group<<<B_ * G_, 256, 0, stream>>>(xyz, cen, pg);

    // 2) T-Net: 6 -> 64 -> 512, maxpool, -> 36, transform
    k_gemm_v<32, 64, 4, false><<<dim3(1, R_ / 128), 256, 0, stream>>>(
        pg, htw1, tn_b1, bufA, nullptr, nullptr, sum_tn1, ssq_tn1);
    k_bnfin<<<1, 64, 0, stream>>>(sum_tn1, ssq_tn1, tn_g1, tn_be1, sc_tn1, sh_tn1, 64, invR);
    k_gemm_v<64, 512, 8, true><<<dim3(4, R_ / 128), 256, 0, stream>>>(
        bufA, htw2, tn_b2, bufB, sc_tn1, sh_tn1, sum_tn2, ssq_tn2);
    k_bnfin<<<2, 256, 0, stream>>>(sum_tn2, ssq_tn2, tn_g2, tn_be2, sc_tn2, sh_tn2, 512, invR);
    k_tnet_maxpool<<<(4096 * 512) / 256, 256, 0, stream>>>(bufB, sc_tn2, sh_tn2, hmax);
    k_tmat<<<(4096 * 36 + 255) / 256, 256, 0, stream>>>(hmax, tn_fw, tn_fb, tmat);
    k_transform<<<R_ / 256, 256, 0, stream>>>(pg, tmat);

    // 3) Main MLP: 10 -> 128 -> 256 -> 512 -> 1024, fused BN+ReLU + stats
    k_gemm_v<32, 128, 8, false><<<dim3(1, R_ / 128), 256, 0, stream>>>(
        pg, hw1, b1, bufA, nullptr, nullptr, sum_l1, ssq_l1);
    k_bnfin<<<1, 128, 0, stream>>>(sum_l1, ssq_l1, g1, be1, sc_l1, sh_l1, 128, invR);
    k_gemm_v<128, 256, 8, true><<<dim3(2, R_ / 128), 256, 0, stream>>>(
        bufA, hw2, b2, bufB, sc_l1, sh_l1, sum_l2, ssq_l2);
    k_bnfin<<<1, 256, 0, stream>>>(sum_l2, ssq_l2, g2, be2, sc_l2, sh_l2, 256, invR);
    k_gemm_v<256, 512, 8, true><<<dim3(4, R_ / 128), 256, 0, stream>>>(
        bufB, hw3, b3, bufA, sc_l2, sh_l2, sum_l3, ssq_l3);
    k_bnfin<<<2, 256, 0, stream>>>(sum_l3, ssq_l3, g3, be3, sc_l3, sh_l3, 512, invR);
    k_gemm_v<512, 1024, 8, true><<<dim3(8, R_ / 128), 256, 0, stream>>>(
        bufA, hw4, b4, bufB, sc_l3, sh_l3, sum_l4, ssq_l4);
    k_bnfin<<<4, 256, 0, stream>>>(sum_l4, ssq_l4, g4, be4, sc_l4, sh_l4, 1024, invR);

    // 4) final BN (no relu) + max over M -> (8,512,1024)
    k_final<<<(4096 * 1024) / 256, 256, 0, stream>>>(bufB, sc_l4, sh_l4, (float*)d_out);
}
